// WaveletLiftingLayer_54219667145431
// MI455X (gfx1250) — matrix operand order repaired and checked by reading, not compiled, not run on hardware
//
#include <hip/hip_runtime.h>
#include <hip/hip_bf16.h>

// ---------------------------------------------------------------------------
// WaveletLiftingLayer on MI455X (gfx1250, wave32, WMMA).
//
//   y[b,c,s] = |  G * ((F * x[b,c] * F) .* H[s]) * G  |
//   F[n,m] = exp(-2*pi*i*n*m/256)      (symmetric)
//   G      = conj(F) / 256             (symmetric, 1/NM folded in)
//
// Precision: matmul operands stored as f16 (hi, lo) pairs; each real product
// uses 3 WMMAs (ah*bh + ah*bl + al*bh) -> ~f32 accuracy at f16 rate / 3.
// Per-image intermediates live in LDS (single f16, 270 KB of the WGP's 320 KB).
//
// NOTE: every k-accumulation loop carries "#pragma unroll 1".  Without it the
// compiler unrolls kc and then LICM-hoists all rb-invariant B fragments
// (~256 VGPRs) out of the rb loop, spilling WMMA operands to scratch.
// ---------------------------------------------------------------------------

typedef __attribute__((ext_vector_type(16))) _Float16 v16h;
typedef __attribute__((ext_vector_type(8)))  _Float16 v8h;
typedef __attribute__((ext_vector_type(8)))  float    v8f;

#define WMMA(a, b, c) \
  __builtin_amdgcn_wmma_f32_16x16x32_f16(false, (a), false, (b), (short)0, (c), false, false)

#define LDT 264   // padded leading dim (halves) for LDS-resident transposed mats

// A operand: 16x32 f16 tile, rows (row0..row0+15) of row-major matrix p.
__device__ __forceinline__ v16h load_a16(const _Float16* p, int ld, int row0,
                                         int k0, int lane) {
  const int l = lane & 15, h = lane >> 4;
  const _Float16* base = p + (size_t)(row0 + l) * ld + k0 + h * 8;
  union { v16h v; v8h q[2]; } u;
  u.q[0] = *(const v8h*)(base);
  u.q[1] = *(const v8h*)(base + 16);
  return u.v;
}

// B operand: 32x16 f16 tile of B[k][n], from COLUMN-MAJOR storage p[n][k]
// (DFT matrices are symmetric, so row-major storage qualifies with ld=256).
__device__ __forceinline__ v16h load_b16_cm(const _Float16* p, int ld, int k0,
                                            int n0, int lane) {
  const int l = lane & 15, h = lane >> 4;
  const _Float16* base = p + (size_t)(n0 + l) * ld + k0 + h * 8;
  union { v16h v; v8h q[2]; } u;
  u.q[0] = *(const v8h*)(base);
  u.q[1] = *(const v8h*)(base + 16);
  return u.v;
}

__device__ __forceinline__ void split_h(float a, _Float16* hi, _Float16* lo) {
  _Float16 h = (_Float16)a;
  *hi = h;
  *lo = (_Float16)(a - (float)h);
}

// ---------------------------------------------------------------------------
// K0: generate split-f16 DFT matrices (forward F, inverse G with 1/256 folded)
// ---------------------------------------------------------------------------
__global__ void dft_gen(_Float16* __restrict__ Frh, _Float16* __restrict__ Frl,
                        _Float16* __restrict__ Fih, _Float16* __restrict__ Fil,
                        _Float16* __restrict__ Grh, _Float16* __restrict__ Grl,
                        _Float16* __restrict__ Gih, _Float16* __restrict__ Gil) {
  const int n = blockIdx.x, m = threadIdx.x;
  const int t = (n * m) & 255;                        // exact phase mod 256
  const float ang = 0.024543692606170259f * (float)t; // 2*pi/256
  const float c = cosf(ang), s = sinf(ang);
  const int idx = n * 256 + m;
  split_h(c,                   &Frh[idx], &Frl[idx]); // F = c - i*s
  split_h(-s,                  &Fih[idx], &Fil[idx]);
  split_h(c * (1.0f / 256.0f), &Grh[idx], &Grl[idx]); // G = (c + i*s)/256
  split_h(s * (1.0f / 256.0f), &Gih[idx], &Gil[idx]);
}

// ---------------------------------------------------------------------------
// K1: forward 2-D DFT of one real image per workgroup.  X = F * x * F.
// ---------------------------------------------------------------------------
__global__ void fwd_fft(const float* __restrict__ x,
                        const _Float16* __restrict__ Frh, const _Float16* __restrict__ Frl,
                        const _Float16* __restrict__ Fih, const _Float16* __restrict__ Fil,
                        float* __restrict__ Xws) {
  extern __shared__ _Float16 smem[];
  _Float16* Tr = smem;                 // 256 * LDT  (T real, transposed)
  _Float16* Ti = Tr + 256 * LDT;       // 256 * LDT  (T imag, transposed)
  _Float16* xh = Ti + 256 * LDT;       // 16 * 256   (row-block of x, hi)
  _Float16* xl = xh + 4096;            // 16 * 256   (row-block of x, lo)

  const int tid = threadIdx.x, wid = tid >> 5, lane = tid & 31;
  const float* xim = x + (size_t)blockIdx.x * 65536;

  // Phase 1: T = x @ F (complex, f16, stored transposed in LDS)
#pragma unroll 1
  for (int rb = 0; rb < 16; ++rb) {
    const int m0 = rb * 16;
    __syncthreads();
    for (int t = tid; t < 4096; t += 256)
      split_h(xim[m0 * 256 + t], &xh[t], &xl[t]);
    __syncthreads();
#pragma unroll 1
    for (int cti = 0; cti < 2; ++cti) {
      const int n0 = (wid + cti * 8) * 16;
      v8f aR = {}, aI = {};
#pragma unroll 1
      for (int kc = 0; kc < 8; ++kc) {
        const int k0 = kc * 32;
        v16h ah  = load_a16(xh, 256, 0, k0, lane);
        v16h al  = load_a16(xl, 256, 0, k0, lane);
        v16h brh = load_b16_cm(Frh, 256, k0, n0, lane);
        v16h brl = load_b16_cm(Frl, 256, k0, n0, lane);
        v16h bih = load_b16_cm(Fih, 256, k0, n0, lane);
        v16h bil = load_b16_cm(Fil, 256, k0, n0, lane);
        aR = WMMA(ah, brh, aR); aR = WMMA(ah, brl, aR); aR = WMMA(al, brh, aR);
        aI = WMMA(ah, bih, aI); aI = WMMA(ah, bil, aI); aI = WMMA(al, bih, aI);
      }
      const int l = lane & 15, hh = lane >> 4;
      for (int r = 0; r < 8; ++r) {   // C/D layout: M = r + 8*hh, N = l
        const int M = m0 + r + 8 * hh, N = n0 + l;
        Tr[N * LDT + M] = (_Float16)aR[r];
        Ti[N * LDT + M] = (_Float16)aI[r];
      }
    }
  }
  __syncthreads();

  // Phase 2: X = F @ T  (f32 to workspace)
  float* Xr = Xws + (size_t)blockIdx.x * 131072;
  float* Xi = Xr + 65536;
#pragma unroll 1
  for (int rb = 0; rb < 16; ++rb) {
    const int m0 = rb * 16;
#pragma unroll 1
    for (int cti = 0; cti < 2; ++cti) {
      const int n0 = (wid + cti * 8) * 16;
      v8f xP = {}, xM = {}, xA = {};   // Xr = P - M, Xi = A
#pragma unroll 1
      for (int kc = 0; kc < 8; ++kc) {
        const int k0 = kc * 32;
        v16h frh = load_a16(Frh, 256, m0, k0, lane);
        v16h frl = load_a16(Frl, 256, m0, k0, lane);
        v16h fih = load_a16(Fih, 256, m0, k0, lane);
        v16h fil = load_a16(Fil, 256, m0, k0, lane);
        v16h btr = load_b16_cm(Tr, LDT, k0, n0, lane);
        v16h bti = load_b16_cm(Ti, LDT, k0, n0, lane);
        xP = WMMA(frh, btr, xP); xP = WMMA(frl, btr, xP);
        xM = WMMA(fih, bti, xM); xM = WMMA(fil, bti, xM);
        xA = WMMA(frh, bti, xA); xA = WMMA(frl, bti, xA);
        xA = WMMA(fih, btr, xA); xA = WMMA(fil, btr, xA);
      }
      const int l = lane & 15, hh = lane >> 4;
      for (int r = 0; r < 8; ++r) {
        const int M = m0 + r + 8 * hh, N = n0 + l;
        Xr[M * 256 + N] = xP[r] - xM[r];
        Xi[M * 256 + N] = xA[r];
      }
    }
  }
}

// ---------------------------------------------------------------------------
// K2: one (b,c,s) image per workgroup: modulate -> inverse 2-D DFT -> |y|.
// ---------------------------------------------------------------------------
__global__ void inv_fft_mod_abs(const float* __restrict__ Xws,
                                const float* __restrict__ Hre, const float* __restrict__ Him,
                                const _Float16* __restrict__ Grh, const _Float16* __restrict__ Grl,
                                const _Float16* __restrict__ Gih, const _Float16* __restrict__ Gil,
                                float* __restrict__ out) {
  extern __shared__ _Float16 smem[];
  _Float16* Bmr = smem;                 // 256 * LDT
  _Float16* Bmi = Bmr + 256 * LDT;      // 256 * LDT
  _Float16* Zrh = Bmi + 256 * LDT;      // 16*256 each: row-block of Z, hi/lo
  _Float16* Zrl = Zrh + 4096;
  _Float16* Zih = Zrl + 4096;
  _Float16* Zil = Zih + 4096;

  const int tid = threadIdx.x, wid = tid >> 5, lane = tid & 31;
  const int bc = blockIdx.x >> 3, s = blockIdx.x & 7;
  const float* Xr = Xws + (size_t)bc * 131072;
  const float* Xi = Xr + 65536;
  const float* hr = Hre + (size_t)s * 65536;
  const float* hi = Him + (size_t)s * 65536;

  // Phase 1: Bm = (X .* H) @ G  (transposed, f16, in LDS)
#pragma unroll 1
  for (int rb = 0; rb < 16; ++rb) {
    const int m0 = rb * 16;
    __syncthreads();
    for (int t = tid; t < 4096; t += 256) {
      const int idx = m0 * 256 + t;
      const float xr = Xr[idx], xi = Xi[idx];
      const float fr = hr[idx], fi = hi[idx];
      split_h(xr * fr - xi * fi, &Zrh[t], &Zrl[t]);
      split_h(xr * fi + xi * fr, &Zih[t], &Zil[t]);
    }
    __syncthreads();
#pragma unroll 1
    for (int cti = 0; cti < 2; ++cti) {
      const int n0 = (wid + cti * 8) * 16;
      v8f bP = {}, bM = {}, bA = {};   // Bmr = P - M, Bmi = A
#pragma unroll 1
      for (int kc = 0; kc < 8; ++kc) {
        const int k0 = kc * 32;
        v16h zrh = load_a16(Zrh, 256, 0, k0, lane);
        v16h zrl = load_a16(Zrl, 256, 0, k0, lane);
        v16h zih = load_a16(Zih, 256, 0, k0, lane);
        v16h zil = load_a16(Zil, 256, 0, k0, lane);
        v16h grh = load_b16_cm(Grh, 256, k0, n0, lane);
        v16h grl = load_b16_cm(Grl, 256, k0, n0, lane);
        v16h gih = load_b16_cm(Gih, 256, k0, n0, lane);
        v16h gil = load_b16_cm(Gil, 256, k0, n0, lane);
        bP = WMMA(zrh, grh, bP); bP = WMMA(zrh, grl, bP); bP = WMMA(zrl, grh, bP);
        bM = WMMA(zih, gih, bM); bM = WMMA(zih, gil, bM); bM = WMMA(zil, gih, bM);
        bA = WMMA(zrh, gih, bA); bA = WMMA(zrh, gil, bA); bA = WMMA(zrl, gih, bA);
        bA = WMMA(zih, grh, bA); bA = WMMA(zih, grl, bA); bA = WMMA(zil, grh, bA);
      }
      const int l = lane & 15, hh = lane >> 4;
      for (int r = 0; r < 8; ++r) {
        const int M = m0 + r + 8 * hh, N = n0 + l;
        Bmr[N * LDT + M] = (_Float16)(bP[r] - bM[r]);
        Bmi[N * LDT + M] = (_Float16)bA[r];
      }
    }
  }
  __syncthreads();

  // Phase 2: y = G @ Bm ; out = |y|
  float* obase = out + (size_t)blockIdx.x * 65536;
#pragma unroll 1
  for (int rb = 0; rb < 16; ++rb) {
    const int m0 = rb * 16;
#pragma unroll 1
    for (int cti = 0; cti < 2; ++cti) {
      const int n0 = (wid + cti * 8) * 16;
      v8f yP = {}, yM = {}, yA = {};
#pragma unroll 1
      for (int kc = 0; kc < 8; ++kc) {
        const int k0 = kc * 32;
        v16h grh = load_a16(Grh, 256, m0, k0, lane);
        v16h grl = load_a16(Grl, 256, m0, k0, lane);
        v16h gih = load_a16(Gih, 256, m0, k0, lane);
        v16h gil = load_a16(Gil, 256, m0, k0, lane);
        v16h br  = load_b16_cm(Bmr, LDT, k0, n0, lane);
        v16h bi  = load_b16_cm(Bmi, LDT, k0, n0, lane);
        yP = WMMA(grh, br, yP); yP = WMMA(grl, br, yP);
        yM = WMMA(gih, bi, yM); yM = WMMA(gil, bi, yM);
        yA = WMMA(grh, bi, yA); yA = WMMA(grl, bi, yA);
        yA = WMMA(gih, br, yA); yA = WMMA(gil, br, yA);
      }
      const int l = lane & 15, hh = lane >> 4;
      for (int r = 0; r < 8; ++r) {
        const int M = m0 + r + 8 * hh, N = n0 + l;
        const float yr = yP[r] - yM[r], yi = yA[r];
        obase[M * 256 + N] = sqrtf(yr * yr + yi * yi);
      }
    }
  }
}

// ---------------------------------------------------------------------------
extern "C" void kernel_launch(void* const* d_in, const int* in_sizes, int n_in,
                              void* d_out, int out_size, void* d_ws, size_t ws_size,
                              hipStream_t stream) {
  (void)in_sizes; (void)n_in; (void)out_size; (void)ws_size;
  const float* x    = (const float*)d_in[0];   // [8,16,256,256] f32
  const float* f_re = (const float*)d_in[1];   // [8,256,256]    f32
  const float* f_im = (const float*)d_in[2];   // [8,256,256]    f32
  float* out = (float*)d_out;                  // [8,16,8,256,256] f32

  // Workspace: 8 half planes (F,G split) = 1 MB, then X (f32) = 64 MB.
  _Float16* w = (_Float16*)d_ws;
  _Float16* Frh = w + 0 * 65536; _Float16* Frl = w + 1 * 65536;
  _Float16* Fih = w + 2 * 65536; _Float16* Fil = w + 3 * 65536;
  _Float16* Grh = w + 4 * 65536; _Float16* Grl = w + 5 * 65536;
  _Float16* Gih = w + 6 * 65536; _Float16* Gil = w + 7 * 65536;
  float* Xws = (float*)(w + 8 * 65536);        // 128 * 131072 floats

  dft_gen<<<256, 256, 0, stream>>>(Frh, Frl, Fih, Fil, Grh, Grl, Gih, Gil);

  const size_t lds1 = (size_t)(2 * 256 * LDT + 2 * 4096) * sizeof(_Float16); // 280 KB
  fwd_fft<<<128, 256, lds1, stream>>>(x, Frh, Frl, Fih, Fil, Xws);

  const size_t lds2 = (size_t)(2 * 256 * LDT + 4 * 4096) * sizeof(_Float16); // 296 KB
  inv_fft_mod_abs<<<1024, 256, lds2, stream>>>(Xws, f_re, f_im,
                                               Grh, Grl, Gih, Gil, out);
}
